// MultiScaleAttention_86320252715349
// MI455X (gfx1250) — matrix operand order repaired and checked by reading, not compiled, not run on hardware
//
#include <hip/hip_runtime.h>

// ---------------------------------------------------------------------------
// MI455X (gfx1250) multi-scale attention, WMMA f16 path, wave32.
// ---------------------------------------------------------------------------

typedef __attribute__((ext_vector_type(16))) _Float16 v16h;
typedef __attribute__((ext_vector_type(8)))  _Float16 v8h;
typedef __attribute__((ext_vector_type(8)))  float    v8f;
typedef __attribute__((ext_vector_type(4)))  int      v4i;

__device__ __forceinline__ v8f wmma16(v16h a, v16h b, v8f c) {
  // v_wmma_f32_16x16x32_f16: D = A(16x32) * B(32x16) + C
  return __builtin_amdgcn_wmma_f32_16x16x32_f16(false, a, false, b, (short)0, c,
                                                false, false);
}

// ---------------------------------------------------------------------------
// Async global->LDS staging (CDNA5 GLOBAL_LOAD_ASYNC_TO_LDS_B128, ASYNCcnt),
// with a portable fallback when the builtin is not declared by the toolchain.
// Signature (from hipcc diagnostic): (v4i __device__* gsrc, v4i __shared__*
// ldst, Ii offset, Ii cpol) -- HIP allows the implicit generic->AS conversion,
// only the pointee type must match.
// ---------------------------------------------------------------------------
#if defined(__AMDGCN__) && __has_builtin(__builtin_amdgcn_global_load_async_to_lds_b128)
#define USE_ASYNC_COPY 1
#else
#define USE_ASYNC_COPY 0
#endif

#if USE_ASYNC_COPY
__device__ __forceinline__ void stage16B(void* lds_dst, const void* gsrc) {
  __builtin_amdgcn_global_load_async_to_lds_b128((v4i*)gsrc, (v4i*)lds_dst, 0, 0);
}
__device__ __forceinline__ void stage_wait() {
  asm volatile("s_wait_asynccnt 0x0" ::: "memory");
}
#else
__device__ __forceinline__ void stage16B(void* lds_dst, const void* gsrc) {
  *(v8h*)lds_dst = *(const v8h*)gsrc;
}
__device__ __forceinline__ void stage_wait() {}  // barrier covers ordering
#endif

// A-matrix fragment (16x32 f16), row-major source, rowStride in halves (mult 8).
// Lane L: row = L%16; 16 contiguous halves at K = 16*(L/16) + e.
__device__ __forceinline__ v16h frag_a(const _Float16* p, int rowStride) {
  const int lane = threadIdx.x & 31;
  const _Float16* q = p + (lane & 15) * rowStride + ((lane >> 4) << 4);
  v16h a;
  ((v8h*)&a)[0] = *(const v8h*)(q);
  ((v8h*)&a)[1] = *(const v8h*)(q + 8);
  return a;
}

// B-matrix fragment (K=32 x N=16) gathered from N-major storage Arr[n][k].
// Lane L: N = L%16, K = 16*(L/16) + e  -> 16 contiguous halves per lane.
__device__ __forceinline__ v16h frag_b_nmajor(const _Float16* p, int rowStride) {
  const int lane = threadIdx.x & 31;
  const _Float16* q = p + (lane & 15) * rowStride + ((lane >> 4) << 4);
  v16h b;
  ((v8h*)&b)[0] = *(const v8h*)(q);
  ((v8h*)&b)[1] = *(const v8h*)(q + 8);
  return b;
}

// B-matrix fragment from K-major storage Arr[k][n] (e.g. a V tile).
__device__ __forceinline__ v16h frag_b_kmajor(const _Float16* p, int rowStride) {
  const int lane = threadIdx.x & 31;
  const int n  = lane & 15;
  const int kb = (lane >> 4) << 4;
  v16h b;
#pragma unroll
  for (int e = 0; e < 16; ++e) b[e] = p[(kb + e) * rowStride + n];
  return b;
}

// ---------------------------------------------------------------------------
// Generic WMMA GEMM:  Out[m][n] = sum_k A[m][k] * W[n][k] + Bias[n]
// Block tile 128x64, BK=32, 256 threads = 8 waves.  Wave w owns the 2x2 tile
// grid { mt = w%4, w%4+4 } x { nt = w/4, w/4+2 }  -> 4 WMMAs per K-step from
// 2 A-fragments + 2 B-fragments (fragment reuse across the 2x2).
// M multiple of 128, N multiple of 64, K multiple of 32.
// ---------------------------------------------------------------------------
template <typename AT, typename OT>
__global__ void __launch_bounds__(256) gemm_wmma(
    const AT* __restrict__ A, int lda,
    const float* __restrict__ W,     // (N,K) row-major, row stride == K
    const float* __restrict__ Bias,  // (N)
    OT* __restrict__ Out, int ldo, int K)
{
  const int m0 = blockIdx.y * 128;
  const int n0 = blockIdx.x * 64;
  const int w  = threadIdx.x >> 5;
  const int lane = threadIdx.x & 31;

  alignas(16) __shared__ _Float16 As[128][40];  // pad 40 -> conflict-free b128
  alignas(16) __shared__ _Float16 Bs[64][40];

  v8f acc[2][2];
#pragma unroll
  for (int i = 0; i < 2; ++i)
#pragma unroll
    for (int j = 0; j < 2; ++j) acc[i][j] = (v8f){};

  const int ra = threadIdx.x >> 1;        // A staging: 0..127
  const int ca = (threadIdx.x & 1) * 16;  // 0,16
  const int rb = threadIdx.x >> 2;        // W staging: 0..63
  const int cb = (threadIdx.x & 3) * 8;   // 0,8,16,24

  for (int k0 = 0; k0 < K; k0 += 32) {
    const AT*    ag = A + (size_t)(m0 + ra) * lda + k0 + ca;
    const float* wg = W + (size_t)(n0 + rb) * K   + k0 + cb;
    if (k0 + 32 < K) {           // gfx1250 global_prefetch_b8 for next K-slab
      __builtin_prefetch(ag + 32, 0, 1);
      __builtin_prefetch(wg + 32, 0, 1);
    }
#pragma unroll
    for (int u = 0; u < 16; ++u) As[ra][ca + u] = (_Float16)ag[u];
#pragma unroll
    for (int u = 0; u < 8; ++u)  Bs[rb][cb + u] = (_Float16)wg[u];
    __syncthreads();

    v16h af0 = frag_a(&As[(w & 3) * 16][0], 40);
    v16h af1 = frag_a(&As[((w & 3) + 4) * 16][0], 40);
    v16h bf0 = frag_b_nmajor(&Bs[(w >> 2) * 16][0], 40);
    v16h bf1 = frag_b_nmajor(&Bs[((w >> 2) + 2) * 16][0], 40);
    acc[0][0] = wmma16(af0, bf0, acc[0][0]);
    acc[0][1] = wmma16(af0, bf1, acc[0][1]);
    acc[1][0] = wmma16(af1, bf0, acc[1][0]);
    acc[1][1] = wmma16(af1, bf1, acc[1][1]);
    __syncthreads();
  }

  const int nl = lane & 15;
  const int hi = lane >> 4;
#pragma unroll
  for (int i = 0; i < 2; ++i) {
    const int mt = (w & 3) + 4 * i;
#pragma unroll
    for (int j = 0; j < 2; ++j) {
      const int nt = (w >> 2) + 2 * j;
      const float bb = Bias[n0 + nt * 16 + nl];
#pragma unroll
      for (int e = 0; e < 8; ++e) {
        const int mrow = m0 + mt * 16 + e + 8 * hi;
        Out[(size_t)mrow * ldo + n0 + nt * 16 + nl] = (OT)(acc[i][j][e] + bb);
      }
    }
  }
}

// ---------------------------------------------------------------------------
// Non-overlapping sequence average-pool of a 256-col slice of an f16 (R,1024)
// tensor into a dense (B*Lk, 256) f16 buffer.
// ---------------------------------------------------------------------------
__global__ void __launch_bounds__(256) pool_kernel(
    const _Float16* __restrict__ src, int colOff,
    _Float16* __restrict__ dst, int s, int Lk)
{
  const int idx = blockIdx.x * 256 + threadIdx.x;  // over B*Lk*256
  const int c   = idx & 255;
  const int row = idx >> 8;                        // b*Lk + j
  const int b   = row / Lk;
  const int j   = row - b * Lk;
  const _Float16* p = src + ((size_t)(b * 2048 + j * s)) * 1024 + colOff + c;
  float acc = 0.f;
  for (int t = 0; t < s; ++t) acc += (float)p[(size_t)t * 1024];
  dst[idx] = (_Float16)(acc * (1.f / (float)s));
}

// ---------------------------------------------------------------------------
// Flash-style attention for one scale.  DS=256, nh=2, hd=128.
//   qh : (B*S, 256) f16     kh, vh : (B*Lk, 256) f16
//   att: (B*S, 256) f16     wmean  : (B, S, Lk) f32 head-mean probs (scale 0)
// Block: 256 threads = 8 waves; wave w -> head w/4, q-row tile 16*(w%4) of a
// 64-row block.  Q fragments in registers; K/V tiles (32 rows) staged in LDS
// via async global->LDS copies.  Pass 1: online softmax + O accumulation.
// Pass 2 (scale 0 only): recompute normalized probs, combine heads through
// LDS, stream mean weights straight to d_out.
// ---------------------------------------------------------------------------
__global__ void __launch_bounds__(256) attn_kernel(
    const _Float16* __restrict__ qh,
    const _Float16* __restrict__ kh,
    const _Float16* __restrict__ vh,
    _Float16* __restrict__ att,
    float* __restrict__ wmean,
    int Lk)
{
  constexpr int S = 2048;
  const int b    = blockIdx.y;
  const int q0   = blockIdx.x * 64;
  const int w    = threadIdx.x >> 5;
  const int lane = threadIdx.x & 31;
  const int mt   = w & 3;
  const int head = w >> 2;
  const int nl   = lane & 15;
  const int hi   = lane >> 4;

  alignas(16) __shared__ _Float16 Ks[32][264];   // K tile, both heads' cols
  alignas(16) __shared__ _Float16 Vs[32][264];   // V tile
  alignas(16) __shared__ _Float16 Ps[8][16][40]; // wave-private P staging
  __shared__ float Wc[4][16][32];                // head-1 probs for pass 2

  // Q fragments: 16 rows x 128 head-dim, 4 depth chunks of 32, kept in VGPRs.
  v16h qf[4];
  {
    const _Float16* qb =
        qh + ((size_t)(b * S + q0 + mt * 16)) * 256 + head * 128;
#pragma unroll
    for (int kc = 0; kc < 4; ++kc) qf[kc] = frag_a(qb + kc * 32, 256);
  }

  float m[8], l[8];
  v8f o[8];
#pragma unroll
  for (int e = 0; e < 8; ++e) { m[e] = -1e30f; l[e] = 0.f; }
#pragma unroll
  for (int t = 0; t < 8; ++t) o[t] = (v8f){};

  const float sc  = 0.08838834764831845f;  // 1/sqrt(128)
  const int   nkb = Lk >> 5;
  const int   lr  = threadIdx.x >> 3;        // staging row 0..31
  const int   lc  = (threadIdx.x & 7) * 32;  // staging col

  for (int kb = 0; kb < nkb; ++kb) {
    {
      const _Float16* kg = kh + ((size_t)(b * Lk + kb * 32 + lr)) * 256 + lc;
      const _Float16* vg = vh + ((size_t)(b * Lk + kb * 32 + lr)) * 256 + lc;
#pragma unroll
      for (int u = 0; u < 4; ++u) {
        stage16B(&Ks[lr][lc + u * 8], kg + u * 8);
        stage16B(&Vs[lr][lc + u * 8], vg + u * 8);
      }
      stage_wait();
    }
    __syncthreads();

    // Scores: 16 q-rows x 32 k-rows = two 16x16 C tiles, depth 128.
    v8f s0 = {}, s1 = {};
#pragma unroll
    for (int kc = 0; kc < 4; ++kc) {
      v16h bk0 = frag_b_nmajor(&Ks[0][head * 128 + kc * 32], 264);
      v16h bk1 = frag_b_nmajor(&Ks[16][head * 128 + kc * 32], 264);
      s0 = wmma16(qf[kc], bk0, s0);
      s1 = wmma16(qf[kc], bk1, s1);
    }

    // Online softmax; C-layout row = e + 8*hi lives in one 16-lane half, so
    // xor masks 1/2/4/8 reduce exactly over that row's 16 columns.
    float p0[8], p1[8];
#pragma unroll
    for (int e = 0; e < 8; ++e) {
      const float a0 = s0[e] * sc, a1 = s1[e] * sc;
      float t = fmaxf(a0, a1);
      t = fmaxf(t, __shfl_xor(t, 1));
      t = fmaxf(t, __shfl_xor(t, 2));
      t = fmaxf(t, __shfl_xor(t, 4));
      t = fmaxf(t, __shfl_xor(t, 8));
      const float mn   = fmaxf(m[e], t);
      const float corr = __expf(m[e] - mn);
      m[e] = mn;
      const float e0 = __expf(a0 - mn), e1 = __expf(a1 - mn);
      p0[e] = e0; p1[e] = e1;
      float rs = e0 + e1;
      rs += __shfl_xor(rs, 1);
      rs += __shfl_xor(rs, 2);
      rs += __shfl_xor(rs, 4);
      rs += __shfl_xor(rs, 8);
      l[e] = l[e] * corr + rs;
#pragma unroll
      for (int t2 = 0; t2 < 8; ++t2) o[t2][e] *= corr;
    }

    // Stage P (C layout -> A layout) via wave-private LDS tile.
#pragma unroll
    for (int e = 0; e < 8; ++e) {
      const int row = e + 8 * hi;
      Ps[w][row][nl]      = (_Float16)p0[e];
      Ps[w][row][16 + nl] = (_Float16)p1[e];
    }
    asm volatile("s_wait_dscnt 0x0" ::: "memory");  // cross-lane LDS round-trip
    v16h pf = frag_a(&Ps[w][0][0], 40);

    // O += P(16x32) @ V(32x128)
#pragma unroll
    for (int t2 = 0; t2 < 8; ++t2) {
      v16h bv = frag_b_kmajor(&Vs[0][head * 128 + t2 * 16], 264);
      o[t2] = wmma16(pf, bv, o[t2]);
    }
    __syncthreads();
  }

  // Normalize and store attended values.
#pragma unroll
  for (int e = 0; e < 8; ++e) l[e] = 1.f / l[e];
#pragma unroll
  for (int t2 = 0; t2 < 8; ++t2) {
#pragma unroll
    for (int e = 0; e < 8; ++e) {
      const int mrow = b * S + q0 + mt * 16 + e + 8 * hi;
      att[(size_t)mrow * 256 + head * 128 + t2 * 16 + nl] =
          (_Float16)(o[t2][e] * l[e]);
    }
  }

  // Pass 2 (scale 0): stream head-mean attention probs straight to d_out.
  if (wmean != nullptr) {
    for (int kb = 0; kb < nkb; ++kb) {
      {
        const _Float16* kg = kh + ((size_t)(b * Lk + kb * 32 + lr)) * 256 + lc;
#pragma unroll
        for (int u = 0; u < 4; ++u) stage16B(&Ks[lr][lc + u * 8], kg + u * 8);
        stage_wait();
      }
      __syncthreads();
      v8f s0 = {}, s1 = {};
#pragma unroll
      for (int kc = 0; kc < 4; ++kc) {
        v16h bk0 = frag_b_nmajor(&Ks[0][head * 128 + kc * 32], 264);
        v16h bk1 = frag_b_nmajor(&Ks[16][head * 128 + kc * 32], 264);
        s0 = wmma16(qf[kc], bk0, s0);
        s1 = wmma16(qf[kc], bk1, s1);
      }
      float p0[8], p1[8];
#pragma unroll
      for (int e = 0; e < 8; ++e) {  // l[] holds 1/l; 0.5 = head mean
        p0[e] = __expf(s0[e] * sc - m[e]) * (0.5f * l[e]);
        p1[e] = __expf(s1[e] * sc - m[e]) * (0.5f * l[e]);
      }
      if (head == 1) {
#pragma unroll
        for (int e = 0; e < 8; ++e) {
          const int row = e + 8 * hi;
          Wc[mt][row][nl]      = p0[e];
          Wc[mt][row][16 + nl] = p1[e];
        }
      }
      __syncthreads();
      if (head == 0) {
#pragma unroll
        for (int e = 0; e < 8; ++e) {
          const int row = e + 8 * hi;
          const size_t base =
              ((size_t)b * S + q0 + mt * 16 + row) * (size_t)Lk + kb * 32;
          wmean[base + nl]      = p0[e] + Wc[mt][row][nl];
          wmean[base + 16 + nl] = p1[e] + Wc[mt][row][16 + nl];
        }
      }
      __syncthreads();
    }
  }
}

// ---------------------------------------------------------------------------
// Host-side orchestration.
// ---------------------------------------------------------------------------
extern "C" void kernel_launch(void* const* d_in, const int* in_sizes, int n_in,
                              void* d_out, int out_size, void* d_ws,
                              size_t ws_size, hipStream_t stream) {
  (void)in_sizes; (void)n_in; (void)out_size; (void)ws_size;

  const float* query = (const float*)d_in[0];
  const float* key_  = (const float*)d_in[1];
  const float* value = (const float*)d_in[2];
  const float* wq    = (const float*)d_in[3];
  const float* bq    = (const float*)d_in[4];
  const float* wk    = (const float*)d_in[5];
  const float* bk    = (const float*)d_in[6];
  const float* wv    = (const float*)d_in[7];
  const float* bv    = (const float*)d_in[8];
  const float* in_w  = (const float*)d_in[9];   // (4, 768, 256)
  const float* in_b  = (const float*)d_in[10];  // (4, 768)
  const float* out_w = (const float*)d_in[11];  // (4, 256, 256)
  const float* out_b = (const float*)d_in[12];  // (4, 256)
  const float* fus_w = (const float*)d_in[13];  // (1024, 1024)
  const float* fus_b = (const float*)d_in[14];  // (1024)

  const int B = 4, S = 2048, H = 1024, DSZ = 256;
  const size_t R = (size_t)B * S;  // 8192 rows
  static const int Lks[4] = {2048, 1024, 512, 256};
  static const int SCL[4] = {1, 2, 4, 8};

  // Workspace carve-up (f16 halves, 32B-aligned chunks).
  _Float16* w16 = (_Float16*)d_ws;
  size_t off = 0;
  auto alloc = [&](size_t n) {
    _Float16* p = w16 + off;
    off += (n + 15) & ~(size_t)15;
    return p;
  };
  _Float16* qF = alloc(R * H);  // projected q/k/v (f16)
  _Float16* kF = alloc(R * H);
  _Float16* vF = alloc(R * H);
  _Float16 *kp[4] = {nullptr, nullptr, nullptr, nullptr};
  _Float16 *vp[4] = {nullptr, nullptr, nullptr, nullptr};
  for (int i = 1; i < 4; ++i) {
    kp[i] = alloc((size_t)B * Lks[i] * DSZ);
    vp[i] = alloc((size_t)B * Lks[i] * DSZ);
  }
  _Float16 *qh[4], *khb[4], *vhb[4], *at[4];
  for (int i = 0; i < 4; ++i) {
    qh[i]  = alloc(R * DSZ);
    khb[i] = alloc((size_t)B * Lks[i] * DSZ);
    vhb[i] = alloc((size_t)B * Lks[i] * DSZ);
    at[i]  = alloc(R * DSZ);
  }
  _Float16* concat = alloc(R * H);

  const dim3 blk(256);

  // 1) QKV projections (f32 in, f16 out).
  {
    dim3 g(H / 64, (unsigned)(R / 128));
    gemm_wmma<float, _Float16><<<g, blk, 0, stream>>>(query, H, wq, bq, qF, H, H);
    gemm_wmma<float, _Float16><<<g, blk, 0, stream>>>(key_,  H, wk, bk, kF, H, H);
    gemm_wmma<float, _Float16><<<g, blk, 0, stream>>>(value, H, wv, bv, vF, H, H);
  }

  // 2) Average-pool K/V slices for scales 2/4/8.
  for (int i = 1; i < 4; ++i) {
    const int total = B * Lks[i] * DSZ;
    pool_kernel<<<total / 256, blk, 0, stream>>>(kF, i * DSZ, kp[i], SCL[i], Lks[i]);
    pool_kernel<<<total / 256, blk, 0, stream>>>(vF, i * DSZ, vp[i], SCL[i], Lks[i]);
  }

  // 3) Per-scale in-projections (f16 GEMMs).
  for (int i = 0; i < 4; ++i) {
    const float* W  = in_w + (size_t)i * 3 * DSZ * DSZ;
    const float* Bi = in_b + (size_t)i * 3 * DSZ;
    {
      dim3 g(DSZ / 64, (unsigned)(R / 128));
      gemm_wmma<_Float16, _Float16><<<g, blk, 0, stream>>>(
          qF + i * DSZ, H, W, Bi, qh[i], DSZ, DSZ);
    }
    const _Float16* ks = (i == 0) ? kF : kp[i];
    const _Float16* vs = (i == 0) ? vF : vp[i];
    const int lda = (i == 0) ? H : DSZ;  // scale0 uses cols [0,256) of kF/vF
    const int Mk  = B * Lks[i];
    dim3 g(DSZ / 64, Mk / 128);
    gemm_wmma<_Float16, _Float16><<<g, blk, 0, stream>>>(
        ks, lda, W + (size_t)DSZ * DSZ, Bi + DSZ, khb[i], DSZ, DSZ);
    gemm_wmma<_Float16, _Float16><<<g, blk, 0, stream>>>(
        vs, lda, W + (size_t)2 * DSZ * DSZ, Bi + 2 * DSZ, vhb[i], DSZ, DSZ);
  }

  // 4) Attention per scale; scale 0 also streams head-mean weights to d_out.
  float* wmean = (float*)d_out + R * H;
  for (int i = 0; i < 4; ++i) {
    dim3 g(S / 64, B);
    attn_kernel<<<g, blk, 0, stream>>>(qh[i], khb[i], vhb[i], at[i],
                                       (i == 0) ? wmean : nullptr, Lks[i]);
  }

  // 5) Out-projections into the concat buffer columns.
  for (int i = 0; i < 4; ++i) {
    dim3 g(DSZ / 64, (unsigned)(R / 128));
    gemm_wmma<_Float16, _Float16><<<g, blk, 0, stream>>>(
        at[i], DSZ, out_w + (size_t)i * DSZ * DSZ, out_b + (size_t)i * DSZ,
        concat + i * DSZ, H, DSZ);
  }

  // 6) Fusion GEMM -> fused f32 output region of d_out.
  {
    dim3 g(H / 64, (unsigned)(R / 128));
    gemm_wmma<_Float16, float><<<g, blk, 0, stream>>>(
        concat, H, fus_w, fus_b, (float*)d_out, H, H);
  }
}
